// Qwen2DecoderLayer_37641093382590
// MI455X (gfx1250) — matrix operand order repaired and checked by reading, not compiled, not run on hardware
//
#include <hip/hip_runtime.h>
#include <hip/hip_bf16.h>

typedef __attribute__((ext_vector_type(16))) _Float16 v16h;
typedef __attribute__((ext_vector_type(8)))  _Float16 v8h;
typedef __attribute__((ext_vector_type(8)))  float    v8f;

#define HIDDEN   3584
#define NHEADS   28
#define HEAD_DIM 128
#define MTOT     8192          // BS * SEQ
#define EPS_F    1e-6f
#define THETA_F  1000000.0f

// ---------------------------------------------------------------------------
// Tensor Data Mover support (gfx1250 device pass only)
// ---------------------------------------------------------------------------
#if defined(__gfx1250__) && __has_builtin(__builtin_amdgcn_tensor_load_to_lds)
#define HAVE_TDM 1
#else
#define HAVE_TDM 0
#endif

#if HAVE_TDM
typedef unsigned int u32x4 __attribute__((ext_vector_type(4)));
typedef int          i32x4 __attribute__((ext_vector_type(4)));
typedef int          i32x8 __attribute__((ext_vector_type(8)));

static __device__ __forceinline__ unsigned lds_off_of(const void* p) {
  // generic -> LDS addrspace -> workgroup-relative byte offset
  return (unsigned)(unsigned long long)
      (const __attribute__((address_space(3))) char*)p;
}

// 2-D tile DMA: global (row-major, 2-byte elements) -> LDS with row padding.
// dims/strides in element (half) units; lds_addr/global addr in bytes.
static __device__ __forceinline__ void
tdm_load_2d_f16(unsigned lds_addr, const void* gptr,
                unsigned tensor_d0, unsigned tensor_d1,
                unsigned tile_d0, unsigned tile_d1,
                unsigned long long stride_d0,
                unsigned pad_interval, unsigned pad_amount) {
  const unsigned long long ga = (unsigned long long)gptr;
  u32x4 g0;
  g0[0] = 1u;                                       // count=1: valid user D#
  g0[1] = lds_addr;                                 // lds_addr (bytes)
  g0[2] = (unsigned)ga;                             // global_addr[31:0]
  g0[3] = (unsigned)((ga >> 32) & 0x1FFFFFFull)     // global_addr[56:32]
        | (2u << 30);                               // type = 2 ("image")
  i32x8 g1;
  g1[0] = (int)((1u << 16)                          // data_size = 2 bytes
        | (1u << 20)                                // pad_enable
        | (pad_interval << 22)
        | (pad_amount << 25));
  g1[1] = (int)((tensor_d0 & 0xFFFFu) << 16);       // tensor_dim0[15:0]
  g1[2] = (int)(((tensor_d0 >> 16) & 0xFFFFu)       // tensor_dim0[31:16]
        | ((tensor_d1 & 0xFFFFu) << 16));           // tensor_dim1[15:0]
  g1[3] = (int)(((tensor_d1 >> 16) & 0xFFFFu)       // tensor_dim1[31:16]
        | ((tile_d0 & 0xFFFFu) << 16));             // tile_dim0
  g1[4] = (int)(tile_d1 & 0xFFFFu);                 // tile_dim1 (tile_dim2=0)
  g1[5] = (int)(stride_d0 & 0xFFFFFFFFull);         // dim0_stride[31:0]
  g1[6] = (int)((stride_d0 >> 32) & 0xFFFFull);     // dim0_stride[47:32]
  g1[7] = 0;                                        // dim1_stride unused (2-D)
  i32x4 z4 = (i32x4)0;
#if __clang_major__ >= 23
  i32x8 z8 = (i32x8)0;
  __builtin_amdgcn_tensor_load_to_lds(g0, g1, z4, z4, z8, 0);
#else
  __builtin_amdgcn_tensor_load_to_lds(g0, g1, z4, z4, 0);
#endif
}

static __device__ __forceinline__ void wait_tensorcnt0() {
#if __has_builtin(__builtin_amdgcn_s_wait_tensorcnt)
  __builtin_amdgcn_s_wait_tensorcnt(0);
#else
  asm volatile("s_wait_tensorcnt 0x0" ::: "memory");
#endif
}
#endif  // HAVE_TDM

// ---------------------------------------------------------------------------
// RMSNorm: x [MTOT][HIDDEN] f32 -> h [MTOT][HIDDEN] f16
// ---------------------------------------------------------------------------
__global__ __launch_bounds__(256)
void rmsnorm_f16_kernel(const float* __restrict__ x,
                        const float* __restrict__ w,
                        _Float16* __restrict__ h) {
  const int row = blockIdx.x;
  const float* xr = x + (size_t)row * HIDDEN;
  __shared__ float red[256];
  float ss = 0.f;
  for (int i = threadIdx.x; i < HIDDEN; i += 256) {
    float v = xr[i];
    ss += v * v;
  }
  red[threadIdx.x] = ss;
  __syncthreads();
  for (int off = 128; off > 0; off >>= 1) {
    if ((int)threadIdx.x < off) red[threadIdx.x] += red[threadIdx.x + off];
    __syncthreads();
  }
  const float scale = rsqrtf(red[0] / (float)HIDDEN + EPS_F);
  _Float16* hr = h + (size_t)row * HIDDEN;
  for (int i = threadIdx.x; i < HIDDEN; i += 256)
    hr[i] = (_Float16)(xr[i] * scale * w[i]);
}

// ---------------------------------------------------------------------------
// Transpose + convert: W [N][K] f32 -> WT [K][N] f16
// ---------------------------------------------------------------------------
__global__ __launch_bounds__(256)
void transpose_f16_kernel(const float* __restrict__ W,
                          _Float16* __restrict__ WT,
                          int N, int K) {
  __shared__ _Float16 tile[32][33];
  const int kb = blockIdx.x * 32;
  const int nb = blockIdx.y * 32;
#pragma unroll
  for (int i = 0; i < 4; ++i) {
    int n = nb + threadIdx.y + i * 8;
    tile[threadIdx.y + i * 8][threadIdx.x] =
        (_Float16)W[(size_t)n * K + kb + threadIdx.x];
  }
  __syncthreads();
#pragma unroll
  for (int i = 0; i < 4; ++i) {
    int k = kb + threadIdx.y + i * 8;
    WT[(size_t)k * N + nb + threadIdx.x] = tile[threadIdx.x][threadIdx.y + i * 8];
  }
}

// ---------------------------------------------------------------------------
// WMMA GEMM:  C[M][N] = A[M][K] (f16 row-major) * BT[K][N] (f16 K-major)
// EPI==0: +bias, store f16.  EPI==1: +resid (f32), store f32.
// 256 threads / 8 waves; tile 128x128x32; wave tile 64x32 (4x2 WMMA tiles).
// Staging via Tensor Data Mover (TDM) when available.
// ---------------------------------------------------------------------------
#define BM 128
#define BN 128
#define BK 32
#define LDA_H 40    // A LDS row stride in halves (64B data + 16B pad)
#define LDB_H 136   // B LDS row stride in halves (256B data + 16B pad)

__device__ __forceinline__ v16h combine16(v8h lo, v8h hi) {
  v16h r;
#pragma unroll
  for (int i = 0; i < 8; ++i) { r[i] = lo[i]; r[i + 8] = hi[i]; }
  return r;
}

template <int EPI>
__global__ __launch_bounds__(256)
void gemm_wmma_kernel(const _Float16* __restrict__ A,
                      const _Float16* __restrict__ BT,
                      const float* __restrict__ bias,
                      const float* __restrict__ resid,
                      void* __restrict__ outp,
                      int M, int N, int K) {
  __shared__ alignas(16) _Float16 lA[2][BM * LDA_H];
  __shared__ alignas(16) _Float16 lB[2][BK * LDB_H];

  const int tid  = threadIdx.x;
  const int lane = tid & 31;
  const int wave = tid >> 5;
  const int wm   = wave & 1;    // wave M offset: wm*64
  const int wn   = wave >> 1;   // wave N offset: wn*32
  const int lg   = lane >> 4;   // lane group (0/1)
  const int lr   = lane & 15;

  const int m0 = blockIdx.y * BM;
  const int n0 = blockIdx.x * BN;
  const int nk = K / BK;   // 112

#if HAVE_TDM
  const unsigned ldsA[2] = { lds_off_of(&lA[0][0]), lds_off_of(&lA[1][0]) };
  const unsigned ldsB[2] = { lds_off_of(&lB[0][0]), lds_off_of(&lB[1][0]) };
  if (wave == 0) {
    // stage 0: A tile rows m0..m0+127, halves 0..31; B tile rows 0..31
    tdm_load_2d_f16(ldsA[0], A + (size_t)m0 * K,
                    (unsigned)K, (unsigned)M, BK, BM,
                    (unsigned long long)K, /*pad 16DW->*/3, /*4DW pad*/3);
    tdm_load_2d_f16(ldsB[0], BT + n0,
                    (unsigned)N, (unsigned)K, BN, BK,
                    (unsigned long long)N, /*64DW->*/5, /*4DW pad*/3);
    wait_tensorcnt0();
  }
#else
  // fallback: per-thread register staging (A: 2 thr/row, B: 8 thr/row)
  const int arow = tid >> 1, acol = (tid & 1) * 16;
  const int brow = tid >> 3, bcol = (tid & 7) * 16;
  const _Float16* gA = A  + (size_t)(m0 + arow) * K + acol;
  const _Float16* gB = BT + (size_t)brow * N + n0 + bcol;
  const int lAoff = arow * LDA_H + acol;
  const int lBoff = brow * LDB_H + bcol;
  v8h ra0 = *(const v8h*)(gA);
  v8h ra1 = *(const v8h*)(gA + 8);
  v8h rb0 = *(const v8h*)(gB);
  v8h rb1 = *(const v8h*)(gB + 8);
  *(v8h*)&lA[0][lAoff]     = ra0;
  *(v8h*)&lA[0][lAoff + 8] = ra1;
  *(v8h*)&lB[0][lBoff]     = rb0;
  *(v8h*)&lB[0][lBoff + 8] = rb1;
#endif
  __syncthreads();

  v8f acc[4][2] = {};

  for (int kb = 0; kb < nk; ++kb) {
    const int cur = kb & 1;

#if HAVE_TDM
    if (wave == 0 && kb + 1 < nk) {
      const int nxt = cur ^ 1;
      tdm_load_2d_f16(ldsA[nxt], A + (size_t)m0 * K + (size_t)(kb + 1) * BK,
                      (unsigned)K, (unsigned)M, BK, BM,
                      (unsigned long long)K, 3, 3);
      tdm_load_2d_f16(ldsB[nxt], BT + (size_t)(kb + 1) * BK * N + n0,
                      (unsigned)N, (unsigned)K, BN, BK,
                      (unsigned long long)N, 5, 3);
    }
#else
    if (kb + 1 < nk) {
      const _Float16* pA = gA + (size_t)(kb + 1) * BK;
      const _Float16* pB = gB + (size_t)(kb + 1) * BK * N;
      ra0 = *(const v8h*)(pA);
      ra1 = *(const v8h*)(pA + 8);
      rb0 = *(const v8h*)(pB);
      rb1 = *(const v8h*)(pB + 8);
    }
#endif

    // ---- fragments from LDS (documented 16-bit WMMA layouts) ----
    v16h af[4], bf[2];
#pragma unroll
    for (int i = 0; i < 4; ++i) {
      const int row = wm * 64 + i * 16 + lr;           // A row m = lane&15
      const _Float16* p = &lA[cur][row * LDA_H + lg * 8];
      v8h lo = *(const v8h*)p;                          // K = g*8 .. g*8+7
      v8h hi = *(const v8h*)(p + 16);                   // K = 16+g*8 .. +7
      af[i] = combine16(lo, hi);
    }
#pragma unroll
    for (int j = 0; j < 2; ++j) {
      const int ncol = wn * 32 + j * 16 + lr;           // B col n = lane&15
#pragma unroll
      for (int s = 0; s < 16; ++s) {
        const int k = (s < 8) ? (lg * 8 + s) : (16 + lg * 8 + (s - 8));
        bf[j][s] = lB[cur][k * LDB_H + ncol];
      }
    }

#pragma unroll
    for (int i = 0; i < 4; ++i)
#pragma unroll
      for (int j = 0; j < 2; ++j)
        acc[i][j] = __builtin_amdgcn_wmma_f32_16x16x32_f16(
            false, af[i], false, bf[j], (short)0, acc[i][j], false, false);

#if HAVE_TDM
    if (wave == 0) wait_tensorcnt0();
#else
    if (kb + 1 < nk) {
      const int nxt = cur ^ 1;
      *(v8h*)&lA[nxt][lAoff]     = ra0;
      *(v8h*)&lA[nxt][lAoff + 8] = ra1;
      *(v8h*)&lB[nxt][lBoff]     = rb0;
      *(v8h*)&lB[nxt][lBoff + 8] = rb1;
    }
#endif
    __syncthreads();
  }

  // ---- epilogue: D layout M = r + 8*lg, N = lr ----
#pragma unroll
  for (int i = 0; i < 4; ++i) {
    const int rowb = m0 + wm * 64 + i * 16 + lg * 8;
#pragma unroll
    for (int j = 0; j < 2; ++j) {
      const int col = n0 + wn * 32 + j * 16 + lr;
      if constexpr (EPI == 0) {
        _Float16* O = (_Float16*)outp;
        const float b = bias[col];
#pragma unroll
        for (int r = 0; r < 8; ++r)
          O[(size_t)(rowb + r) * N + col] = (_Float16)(acc[i][j][r] + b);
      } else {
        float* O = (float*)outp;
#pragma unroll
        for (int r = 0; r < 8; ++r) {
          const size_t idx = (size_t)(rowb + r) * N + col;
          O[idx] = acc[i][j][r] + resid[idx];
        }
      }
    }
  }
}

// ---------------------------------------------------------------------------
// RoPE (neox) in place on q [MTOT][NHEADS*HEAD_DIM] f16
// ---------------------------------------------------------------------------
__global__ __launch_bounds__(256)
void rope_kernel(_Float16* __restrict__ q, const int* __restrict__ positions) {
  const int idx = blockIdx.x * 256 + threadIdx.x;
  const int j  = idx & 63;
  const int t  = idx >> 6;
  const int hd = t % NHEADS;
  const int m  = t / NHEADS;
  if (m >= MTOT) return;
  const float pos = (float)positions[m];
  const float inv = expf(-(2.0f * (float)j / (float)HEAD_DIM) * logf(THETA_F));
  const float ang = pos * inv;
  const float c = cosf(ang), s = sinf(ang);
  const size_t base = (size_t)m * HIDDEN + (size_t)hd * HEAD_DIM;
  const float x1 = (float)q[base + j];
  const float x2 = (float)q[base + 64 + j];
  q[base + j]      = (_Float16)(x1 * c - x2 * s);
  q[base + 64 + j] = (_Float16)(x2 * c + x1 * s);
}

// ---------------------------------------------------------------------------
// Launcher  (inputs: 0:x 1:positions 2:ln_w 3:Wq 4:bq 5:Wk 6:bk 7:Wv 8:bv 9:Wo)
// k/v paths are dead code in the reference -> skipped.
// ---------------------------------------------------------------------------
extern "C" void kernel_launch(void* const* d_in, const int* in_sizes, int n_in,
                              void* d_out, int out_size, void* d_ws, size_t ws_size,
                              hipStream_t stream) {
  (void)in_sizes; (void)n_in; (void)out_size; (void)ws_size;

  const float* x    = (const float*)d_in[0];
  const int*   pos  = (const int*)d_in[1];
  const float* ln_w = (const float*)d_in[2];
  const float* Wq   = (const float*)d_in[3];
  const float* bq   = (const float*)d_in[4];
  const float* Wo   = (const float*)d_in[9];

  constexpr size_t H_BYTES = (size_t)MTOT * HIDDEN * sizeof(_Float16);
  constexpr size_t W_BYTES = (size_t)HIDDEN * HIDDEN * sizeof(_Float16);

  char* ws = (char*)d_ws;
  _Float16* h   = (_Float16*)(ws);
  _Float16* WqT = (_Float16*)(ws + H_BYTES);
  _Float16* WoT = (_Float16*)(ws + H_BYTES + W_BYTES);
  _Float16* q   = (_Float16*)(ws + H_BYTES + 2 * W_BYTES);

  rmsnorm_f16_kernel<<<MTOT, 256, 0, stream>>>(x, ln_w, h);

  dim3 tb(32, 8);
  dim3 tg(HIDDEN / 32, HIDDEN / 32);
  transpose_f16_kernel<<<tg, tb, 0, stream>>>(Wq, WqT, HIDDEN, HIDDEN);
  transpose_f16_kernel<<<tg, tb, 0, stream>>>(Wo, WoT, HIDDEN, HIDDEN);

  dim3 gg(HIDDEN / BN, MTOT / BM);  // 28 x 64
  gemm_wmma_kernel<0><<<gg, 256, 0, stream>>>(h, WqT, bq, nullptr, (void*)q,
                                              MTOT, HIDDEN, HIDDEN);

  const int rope_threads = MTOT * NHEADS * 64;
  rope_kernel<<<rope_threads / 256, 256, 0, stream>>>(q, pos);

  gemm_wmma_kernel<1><<<gg, 256, 0, stream>>>(q, WoT, nullptr, x, d_out,
                                              MTOT, HIDDEN, HIDDEN);
}
